// ReachabilityFeaturesGNN_85925115724411
// MI455X (gfx1250) — matrix operand order repaired and checked by reading, not compiled, not run on hardware
//
#include <hip/hip_runtime.h>

typedef __attribute__((ext_vector_type(16))) _Float16 v16h;
typedef __attribute__((ext_vector_type(8)))  _Float16 v8h;
typedef __attribute__((ext_vector_type(8)))  float    v8f;

#define LEAKY 0.2f
#define BN_EPS 1e-5f

__device__ __forceinline__ unsigned flipf(float f) {
  unsigned u = __float_as_uint(f);
  return (u >> 31) ? ~u : (u | 0x80000000u);
}
__device__ __forceinline__ float unflipf(unsigned u) {
  u = (u >> 31) ? (u & 0x7FFFFFFFu) : ~u;
  return __uint_as_float(u);
}

// ---------------- utility fills ----------------
__global__ void k_fill_f32(float* p, float v, int n) {
  int i = blockIdx.x * blockDim.x + threadIdx.x;
  if (i < n) p[i] = v;
}
__global__ void k_fill_u32(unsigned* p, unsigned v, int n) {
  int i = blockIdx.x * blockDim.x + threadIdx.x;
  if (i < n) p[i] = v;
}

// ---------------- self-loop edge_attr mean (PyG fill_value='mean') ----------------
__global__ void k_ea_scatter(const int* __restrict__ dst, const float* __restrict__ eattr,
                             float* __restrict__ sum, float* __restrict__ cnt, int E0) {
  int e = blockIdx.x * blockDim.x + threadIdx.x;
  if (e >= E0) return;
  int d = dst[e];
  atomicAdd(&cnt[d], 1.f);
  #pragma unroll
  for (int f = 0; f < 4; ++f) atomicAdd(&sum[(size_t)d * 4 + f], eattr[(size_t)e * 4 + f]);
}
__global__ void k_ea_mean(float* __restrict__ sum, const float* __restrict__ cnt, int N) {
  int i = blockIdx.x * blockDim.x + threadIdx.x;
  if (i >= N) return;
  float c = fmaxf(cnt[i], 1.f);
  #pragma unroll
  for (int f = 0; f < 4; ++f) sum[(size_t)i * 4 + f] /= c;
}

// ---------------- pack A: f32 [M,K] -> zero-padded f16 [Mp,Kp], row-major ----------------
// one thread per 8 contiguous output halves (Kp % 32 == 0, so chunks never straddle rows)
__global__ void k_pack_a(const float* __restrict__ A, _Float16* __restrict__ Ap,
                         int M, int K, int Kp, int chunks) {
  int idx = blockIdx.x * blockDim.x + threadIdx.x;
  if (idx >= chunks) return;
  int r  = (idx * 8) / Kp;
  int k0 = (idx * 8) % Kp;
  v8h v;
  #pragma unroll
  for (int t = 0; t < 8; ++t) {
    int k = k0 + t;
    v[t] = (_Float16)((r < M && k < K) ? A[(size_t)r * K + k] : 0.f);
  }
  *(v8h*)(Ap + (size_t)idx * 8) = v;
}

// ---------------- pack B: f32 [K,N] -> WMMA B-fragment layout ----------------
// layout: Bp[tn][ks][lane][16 halves], fragment half t holds B[ks*32 + 16*(lane>>4) + t][col]
__global__ void k_pack_b(const float* __restrict__ B, _Float16* __restrict__ Bp,
                         int K, int N, int kSteps, int nT) {
  int idx = blockIdx.x * blockDim.x + threadIdx.x;
  int total = nT * kSteps * 32;
  if (idx >= total) return;
  int lane = idx & 31;
  int ks   = (idx >> 5) % kSteps;
  int tn   = idx / (32 * kSteps);
  int half = lane >> 4;
  int col  = tn * 16 + (lane & 15);
  v8h lo, hi;
  #pragma unroll
  for (int t = 0; t < 8; ++t) {
    int k0 = ks * 32 + 16 * half + t;
    int k1 = k0 + 8;
    lo[t] = (_Float16)((col < N && k0 < K) ? B[(size_t)k0 * N + col] : 0.f);
    hi[t] = (_Float16)((col < N && k1 < K) ? B[(size_t)k1 * N + col] : 0.f);
  }
  _Float16* outp = Bp + (size_t)idx * 16;
  *(v8h*)(outp)     = lo;
  *(v8h*)(outp + 8) = hi;
}

// ---------------- WMMA GEMM on pre-packed operands ----------------
// C[M,N] = act(A[M,K] @ B[K,N] + bias); Ap padded f16 [Mp,Kp]; Bp fragment-packed.
// One wave per 16x16 tile; inner loop: 4x global_load_b128 + 1x v_wmma.
__global__ void __launch_bounds__(256)
k_wmma_gemm(const _Float16* __restrict__ Ap, const _Float16* __restrict__ Bp,
            const float* __restrict__ bias, float* __restrict__ C,
            int M, int N, int Kp, int relu) {
  int lane = threadIdx.x & 31;
  int tile = blockIdx.x * (blockDim.x >> 5) + (threadIdx.x >> 5);
  int mT = (M + 15) >> 4, nT = (N + 15) >> 4;
  if (tile >= mT * nT) return;               // wave-uniform: EXEC all-1s inside
  int tm = tile / nT, tn = tile % nT;
  int half = lane >> 4;
  int row = tm * 16 + (lane & 15);           // always < Mp (Ap is padded)
  int col = tn * 16 + (lane & 15);
  int kSteps = Kp >> 5;
  const _Float16* Arow = Ap + (size_t)row * Kp + 16 * half;
  const _Float16* Bfrag = Bp + ((size_t)tn * kSteps * 32 + lane) * 16;
  v8f acc = {0.f, 0.f, 0.f, 0.f, 0.f, 0.f, 0.f, 0.f};
  for (int ks = 0; ks < kSteps; ++ks) {
    v8h alo = *(const v8h*)(Arow);           // K = kb+16h .. +7
    v8h ahi = *(const v8h*)(Arow + 8);       // K = kb+16h+8 .. +7
    v8h blo = *(const v8h*)(Bfrag);
    v8h bhi = *(const v8h*)(Bfrag + 8);
    v16h a = __builtin_shufflevector(alo, ahi, 0,1,2,3,4,5,6,7,8,9,10,11,12,13,14,15);
    v16h b = __builtin_shufflevector(blo, bhi, 0,1,2,3,4,5,6,7,8,9,10,11,12,13,14,15);
    acc = __builtin_amdgcn_wmma_f32_16x16x32_f16(false, a, false, b, (short)0, acc,
                                                 false, false);
    Arow  += 32;
    Bfrag += 32 * 16;
  }
  float bcol = (bias != nullptr && col < N) ? bias[col] : 0.f;
  // C/D layout: VGPR j -> M = j + 8*half, N = lane%16
  #pragma unroll
  for (int j = 0; j < 8; ++j) {
    int r = tm * 16 + 8 * half + j;
    if (r < M && col < N) {
      float v = acc[j] + bcol;
      if (relu) v = fmaxf(v, 0.f);
      C[(size_t)r * N + col] = v;
    }
  }
}

// ---------------- GATv2 edge logits + segment max (wave per edge) ----------------
__global__ void k_gat_logits(const float* __restrict__ xl, const float* __restrict__ xr,
                             const int* __restrict__ src0, const int* __restrict__ dst0,
                             const float* __restrict__ eattr, const float* __restrict__ mean_ea,
                             const float* __restrict__ We, const float* __restrict__ att,
                             float* __restrict__ logi, unsigned* __restrict__ mx,
                             int E0, int N, int H) {
  int e = blockIdx.x * (blockDim.x >> 5) + (threadIdx.x >> 5);
  int Etot = E0 + N;
  if (e >= Etot) return;
  int lane = threadIdx.x & 31;
  int s, d;
  const float* ea;
  if (e < E0) { s = src0[e]; d = dst0[e]; ea = eattr + (size_t)e * 4; }
  else        { s = e - E0;  d = s;       ea = mean_ea + (size_t)s * 4; }
  float e0 = ea[0], e1 = ea[1], e2 = ea[2], e3 = ea[3];
  int F = H * 64;
  const float* xls = xl + (size_t)s * F;
  const float* xrd = xr + (size_t)d * F;
  for (int h = 0; h < H; ++h) {
    float acc = 0.f;
    #pragma unroll
    for (int cc = 0; cc < 64; cc += 32) {
      int c = h * 64 + cc + lane;
      float w = e0 * We[c] + e1 * We[F + c] + e2 * We[2 * F + c] + e3 * We[3 * F + c];
      float v = xls[c] + xrd[c] + w;
      v = (v > 0.f) ? v : LEAKY * v;
      acc += v * att[c];
    }
    #pragma unroll
    for (int o = 16; o > 0; o >>= 1) acc += __shfl_xor(acc, o, 32);
    if (lane == 0) {
      logi[(size_t)e * H + h] = acc;
      atomicMax(&mx[(size_t)d * H + h], flipf(acc));
    }
  }
}

// ---------------- exp(logit - max) and segment sum ----------------
__global__ void k_gat_expden(const int* __restrict__ dst0, float* __restrict__ logi,
                             const unsigned* __restrict__ mx, float* __restrict__ den,
                             int E0, int N, int H) {
  int idx = blockIdx.x * blockDim.x + threadIdx.x;
  int Etot = E0 + N;
  if (idx >= Etot * H) return;
  int e = idx / H, h = idx - e * H;
  int d = (e < E0) ? dst0[e] : (e - E0);
  float m = unflipf(mx[(size_t)d * H + h]);
  float ex = __expf(logi[idx] - m);
  logi[idx] = ex;   // reuse buffer for exp values
  atomicAdd(&den[(size_t)d * H + h], ex);
}

// ---------------- alpha-weighted scatter aggregation (wave per edge) ----------------
__global__ void k_gat_aggregate(const float* __restrict__ xl, const int* __restrict__ src0,
                                const int* __restrict__ dst0, const float* __restrict__ ex,
                                const float* __restrict__ den, float* __restrict__ out,
                                int E0, int N, int H) {
  int e = blockIdx.x * (blockDim.x >> 5) + (threadIdx.x >> 5);
  int Etot = E0 + N;
  if (e >= Etot) return;
  int lane = threadIdx.x & 31;
  int s, d;
  if (e < E0) { s = src0[e]; d = dst0[e]; }
  else        { s = e - E0;  d = s; }
  int F = H * 64;
  const float* xls = xl + (size_t)s * F;
  float* outd = out + (size_t)d * F;
  for (int h = 0; h < H; ++h) {
    float a = ex[(size_t)e * H + h] / (den[(size_t)d * H + h] + 1e-16f);
    #pragma unroll
    for (int cc = 0; cc < 64; cc += 32) {
      int c = h * 64 + cc + lane;
      atomicAdd(&outd[c], a * xls[c]);
    }
  }
}

// ---------------- BatchNorm stats (block per channel) + apply(+ReLU) ----------------
__global__ void __launch_bounds__(256)
k_bn_stats(const float* __restrict__ h, int Nn, int F, float* __restrict__ mu,
           float* __restrict__ rs) {
  int c = blockIdx.x;
  float s = 0.f, s2 = 0.f;
  for (int i = threadIdx.x; i < Nn; i += blockDim.x) {
    float v = h[(size_t)i * F + c];
    s += v; s2 += v * v;
  }
  __shared__ float sh[512];
  sh[threadIdx.x] = s; sh[256 + threadIdx.x] = s2;
  __syncthreads();
  for (int o = 128; o > 0; o >>= 1) {
    if ((int)threadIdx.x < o) {
      sh[threadIdx.x] += sh[threadIdx.x + o];
      sh[256 + threadIdx.x] += sh[256 + threadIdx.x + o];
    }
    __syncthreads();
  }
  if (threadIdx.x == 0) {
    float m = sh[0] / (float)Nn;
    float var = sh[256] / (float)Nn - m * m;
    mu[c] = m;
    rs[c] = rsqrtf(var + BN_EPS);
  }
}
__global__ void k_bn_apply(float* __restrict__ h, const float* __restrict__ mu,
                           const float* __restrict__ rs, const float* __restrict__ g,
                           const float* __restrict__ be, int total, int F) {
  int idx = blockIdx.x * blockDim.x + threadIdx.x;
  if (idx >= total) return;
  int c = idx % F;
  float v = (h[idx] - mu[c]) * rs[c] * g[c] + be[c];
  h[idx] = fmaxf(v, 0.f);
}

// ---------------- concat [h2 | ce[batch]] ----------------
__global__ void k_concat(const float* __restrict__ h2, const float* __restrict__ ce,
                         const int* __restrict__ batch, float* __restrict__ fused, int Nn) {
  int idx = blockIdx.x * blockDim.x + threadIdx.x;
  if (idx >= Nn * 128) return;
  int i = idx >> 7, j = idx & 127;
  fused[idx] = (j < 64) ? h2[(size_t)i * 64 + j]
                        : ce[(size_t)batch[i] * 64 + (j - 64)];
}

extern "C" void kernel_launch(void* const* d_in, const int* in_sizes, int n_in,
                              void* d_out, int out_size, void* d_ws, size_t ws_size,
                              hipStream_t stream) {
  const float* x     = (const float*)d_in[0];
  const int*   esrc  = (const int*)d_in[1];
  const int*   edst  = (const int*)d_in[2];
  const float* eat   = (const float*)d_in[3];
  const int*   batch = (const int*)d_in[4];
  const float* clim  = (const float*)d_in[5];
  const float* Wl1   = (const float*)d_in[6];
  const float* Wr1   = (const float*)d_in[7];
  const float* We1   = (const float*)d_in[8];
  const float* att1  = (const float*)d_in[9];
  // d_in[10] = b1, d_in[17] = b2: per-channel biases before BatchNorm cancel exactly.
  const float* g1    = (const float*)d_in[11];
  const float* be1   = (const float*)d_in[12];
  const float* Wl2   = (const float*)d_in[13];
  const float* Wr2   = (const float*)d_in[14];
  const float* We2   = (const float*)d_in[15];
  const float* att2  = (const float*)d_in[16];
  const float* g2    = (const float*)d_in[18];
  const float* be2   = (const float*)d_in[19];
  const float* Wc    = (const float*)d_in[20];
  const float* bc    = (const float*)d_in[21];
  const float* Wcl1  = (const float*)d_in[22];
  const float* bcl1  = (const float*)d_in[23];
  const float* Wcl2  = (const float*)d_in[24];
  const float* bcl2  = (const float*)d_in[25];
  (void)n_in; (void)out_size; (void)ws_size;

  const int Nn = in_sizes[0] / 10;
  const int E0 = in_sizes[1];
  const int G  = in_sizes[5] / 6;
  const int Etot = E0 + Nn;

  auto cdiv = [](int a, int b) { return (a + b - 1) / b; };
  auto pad16 = [&](int v) { return cdiv(v, 16) * 16; };
  auto pad32 = [&](int v) { return cdiv(v, 32) * 32; };

  // ---- workspace layout (byte arena, 256B aligned blocks) ----
  char* base = (char*)d_ws;
  size_t off = 0;
  auto alloc = [&](size_t bytes) -> void* {
    void* p = base + off;
    off += (bytes + 255) & ~(size_t)255;
    return p;
  };
  float*    mean_ea = (float*)alloc((size_t)Nn * 4 * 4);
  float*    cnt     = (float*)alloc((size_t)Nn * 4);
  float*    xl1     = (float*)alloc((size_t)Nn * 256 * 4);
  float*    xr1     = (float*)alloc((size_t)Nn * 256 * 4);
  float*    log1    = (float*)alloc((size_t)Etot * 4 * 4);
  unsigned* mx1     = (unsigned*)alloc((size_t)Nn * 4 * 4);
  float*    den1    = (float*)alloc((size_t)Nn * 4 * 4);
  float*    h1      = (float*)alloc((size_t)Nn * 256 * 4);
  float*    mu1     = (float*)alloc(256 * 4);
  float*    rs1     = (float*)alloc(256 * 4);
  float*    xl2     = (float*)alloc((size_t)Nn * 64 * 4);
  float*    xr2     = (float*)alloc((size_t)Nn * 64 * 4);
  float*    log2    = (float*)alloc((size_t)Etot * 4);
  unsigned* mx2     = (unsigned*)alloc((size_t)Nn * 4);
  float*    den2    = (float*)alloc((size_t)Nn * 4);
  float*    h2      = (float*)alloc((size_t)Nn * 64 * 4);
  float*    mu2     = (float*)alloc(64 * 4);
  float*    rs2     = (float*)alloc(64 * 4);
  float*    ce      = (float*)alloc((size_t)G * 64 * 4);
  float*    fused   = (float*)alloc((size_t)Nn * 128 * 4);
  float*    hidden  = (float*)alloc((size_t)Nn * 64 * 4);
  // packed f16 activations
  _Float16* xP      = (_Float16*)alloc((size_t)pad16(Nn) * pad32(10) * 2);
  _Float16* h1P     = (_Float16*)alloc((size_t)pad16(Nn) * 256 * 2);
  _Float16* climP   = (_Float16*)alloc((size_t)pad16(G) * pad32(6) * 2);
  _Float16* fusedP  = (_Float16*)alloc((size_t)pad16(Nn) * 128 * 2);
  _Float16* hiddenP = (_Float16*)alloc((size_t)pad16(Nn) * 64 * 2);
  // packed f16 weights (fragment layout): nT*kSteps*512 halves
  auto bpackBytes = [&](int K, int N) { return (size_t)cdiv(N, 16) * cdiv(K, 32) * 512 * 2; };
  _Float16* Wl1P  = (_Float16*)alloc(bpackBytes(10, 256));
  _Float16* Wr1P  = (_Float16*)alloc(bpackBytes(10, 256));
  _Float16* Wl2P  = (_Float16*)alloc(bpackBytes(256, 64));
  _Float16* Wr2P  = (_Float16*)alloc(bpackBytes(256, 64));
  _Float16* WcP   = (_Float16*)alloc(bpackBytes(6, 64));
  _Float16* Wcl1P = (_Float16*)alloc(bpackBytes(128, 64));
  _Float16* Wcl2P = (_Float16*)alloc(bpackBytes(64, 4));

  auto packA = [&](const float* A, _Float16* Ap, int M, int K) {
    int chunks = pad16(M) * pad32(K) / 8;
    k_pack_a<<<cdiv(chunks, 256), 256, 0, stream>>>(A, Ap, M, K, pad32(K), chunks);
  };
  auto packB = [&](const float* B, _Float16* Bp, int K, int N) {
    int total = cdiv(N, 16) * cdiv(K, 32) * 32;
    k_pack_b<<<cdiv(total, 256), 256, 0, stream>>>(B, Bp, K, N, cdiv(K, 32), cdiv(N, 16));
  };
  auto gemm = [&](const _Float16* Ap, const _Float16* Bp, const float* bias, float* C,
                  int M, int N, int K, int relu) {
    int tiles = cdiv(M, 16) * cdiv(N, 16);
    k_wmma_gemm<<<cdiv(tiles, 8), 256, 0, stream>>>(Ap, Bp, bias, C, M, N, pad32(K), relu);
  };
  const unsigned NEG_INF_FLIP = 0x00800000u;  // flip(-FLT_MAX)

  // ---- pack all weights once ----
  packB(Wl1, Wl1P, 10, 256);
  packB(Wr1, Wr1P, 10, 256);
  packB(Wl2, Wl2P, 256, 64);
  packB(Wr2, Wr2P, 256, 64);
  packB(Wc, WcP, 6, 64);
  packB(Wcl1, Wcl1P, 128, 64);
  packB(Wcl2, Wcl2P, 64, 4);

  // ---- self-loop edge_attr mean ----
  k_fill_f32<<<cdiv(Nn * 4, 256), 256, 0, stream>>>(mean_ea, 0.f, Nn * 4);
  k_fill_f32<<<cdiv(Nn, 256), 256, 0, stream>>>(cnt, 0.f, Nn);
  k_ea_scatter<<<cdiv(E0, 256), 256, 0, stream>>>(edst, eat, mean_ea, cnt, E0);
  k_ea_mean<<<cdiv(Nn, 256), 256, 0, stream>>>(mean_ea, cnt, Nn);

  // ---- GATv2 layer 1 (H=4, C=64) ----
  packA(x, xP, Nn, 10);
  gemm(xP, Wl1P, nullptr, xl1, Nn, 256, 10, 0);
  gemm(xP, Wr1P, nullptr, xr1, Nn, 256, 10, 0);
  k_fill_u32<<<cdiv(Nn * 4, 256), 256, 0, stream>>>(mx1, NEG_INF_FLIP, Nn * 4);
  k_fill_f32<<<cdiv(Nn * 4, 256), 256, 0, stream>>>(den1, 0.f, Nn * 4);
  k_fill_f32<<<cdiv(Nn * 256, 256), 256, 0, stream>>>(h1, 0.f, Nn * 256);
  k_gat_logits<<<cdiv(Etot, 8), 256, 0, stream>>>(xl1, xr1, esrc, edst, eat, mean_ea,
                                                  We1, att1, log1, mx1, E0, Nn, 4);
  k_gat_expden<<<cdiv(Etot * 4, 256), 256, 0, stream>>>(edst, log1, mx1, den1, E0, Nn, 4);
  k_gat_aggregate<<<cdiv(Etot, 8), 256, 0, stream>>>(xl1, esrc, edst, log1, den1, h1,
                                                     E0, Nn, 4);
  k_bn_stats<<<256, 256, 0, stream>>>(h1, Nn, 256, mu1, rs1);
  k_bn_apply<<<cdiv(Nn * 256, 256), 256, 0, stream>>>(h1, mu1, rs1, g1, be1, Nn * 256, 256);

  // ---- GATv2 layer 2 (H=1, C=64) ----
  packA(h1, h1P, Nn, 256);
  gemm(h1P, Wl2P, nullptr, xl2, Nn, 64, 256, 0);
  gemm(h1P, Wr2P, nullptr, xr2, Nn, 64, 256, 0);
  k_fill_u32<<<cdiv(Nn, 256), 256, 0, stream>>>(mx2, NEG_INF_FLIP, Nn);
  k_fill_f32<<<cdiv(Nn, 256), 256, 0, stream>>>(den2, 0.f, Nn);
  k_fill_f32<<<cdiv(Nn * 64, 256), 256, 0, stream>>>(h2, 0.f, Nn * 64);
  k_gat_logits<<<cdiv(Etot, 8), 256, 0, stream>>>(xl2, xr2, esrc, edst, eat, mean_ea,
                                                  We2, att2, log2, mx2, E0, Nn, 1);
  k_gat_expden<<<cdiv(Etot, 256), 256, 0, stream>>>(edst, log2, mx2, den2, E0, Nn, 1);
  k_gat_aggregate<<<cdiv(Etot, 8), 256, 0, stream>>>(xl2, esrc, edst, log2, den2, h2,
                                                     E0, Nn, 1);
  k_bn_stats<<<64, 256, 0, stream>>>(h2, Nn, 64, mu2, rs2);
  k_bn_apply<<<cdiv(Nn * 64, 256), 256, 0, stream>>>(h2, mu2, rs2, g2, be2, Nn * 64, 64);

  // ---- climber MLP + head ----
  packA(clim, climP, G, 6);
  gemm(climP, WcP, bc, ce, G, 64, 6, 1);
  k_concat<<<cdiv(Nn * 128, 256), 256, 0, stream>>>(h2, ce, batch, fused, Nn);
  packA(fused, fusedP, Nn, 128);
  gemm(fusedP, Wcl1P, bcl1, hidden, Nn, 64, 128, 1);
  packA(hidden, hiddenP, Nn, 64);
  gemm(hiddenP, Wcl2P, bcl2, (float*)d_out, Nn, 4, 64, 0);
}
